// GCN_one_graph_15350213116759
// MI455X (gfx1250) — matrix operand order repaired and checked by reading, not compiled, not run on hardware
//
#include <hip/hip_runtime.h>
#include <hip/hip_bf16.h>
#include <math.h>

// ---------------------------------------------------------------------------
// GCN (5-hop, HID=128) for MI455X / gfx1250.
// GEMMs: v_wmma_f32_16x16x32_bf16 (bf16 in, f32 accum). Scatter: f32 atomics.
// ---------------------------------------------------------------------------

#define NN      100000
#define NE      1600000
#define HID     128
#define ODIM    64
#define NHOPS   5
#define LN_EPS  1e-5f

typedef __attribute__((ext_vector_type(16))) __bf16 bf16x16;
typedef __attribute__((ext_vector_type(8)))  float  f32x8;
typedef __attribute__((ext_vector_type(4)))  float  f32x4;

// ---------------------------------------------------------------------------
// Repack a fp32 weight [128 x ncols] into bf16 B-fragments in per-lane order:
// packed[((t*4 + kk)*32 + lane)*16 + e], lane<16 -> K {0..7,16..23}, lane>=16 -> K {8..15,24..31}.
// ---------------------------------------------------------------------------
__global__ void pack_weight(const float* __restrict__ W, __bf16* __restrict__ dst,
                            int ncols) {
  int ntiles = ncols >> 4;
  int total  = ntiles * 2048;                 // 4 kk * 32 lanes * 16 elems
  int p = blockIdx.x * blockDim.x + threadIdx.x;
  if (p >= total) return;
  int e    = p & 15;
  int lane = (p >> 4) & 31;
  int kk   = (p >> 9) & 3;
  int t    = p >> 11;
  int col  = t * 16 + (lane & 15);
  int k    = kk * 32 + ((lane < 16) ? 0 : 8) + (e & 7) + ((e >> 3) << 4);   // B: lanes 0-15 K={0..7,16..23}, 16-31 K={8..15,24..31}
  dst[p] = (__bf16)W[k * ncols + col];
}

// ---------------------------------------------------------------------------
// WMMA GEMM: Y[N, NT*16] = act(X[N,128] @ Wpacked + bias) (+ residual)
// One wave computes a 16-row x (NT*16)-col slab. K=128 = 4 x K32 WMMA steps.
// ---------------------------------------------------------------------------
template <int NT, bool GELU, bool HAS_BIAS, bool HAS_RES>
__global__ __launch_bounds__(256) void gemm_bf16(
    const float* __restrict__ X, const __bf16* __restrict__ Wp,
    const float* __restrict__ bias, const float* __restrict__ res,
    float* __restrict__ Y, int nrows) {
  const int lane    = threadIdx.x & 31;
  const int wave    = (blockIdx.x * blockDim.x + threadIdx.x) >> 5;
  const int node0   = wave * 16;
  if (node0 >= nrows) return;                 // uniform per wave: EXEC stays all-1s
  const int halfsel = lane >> 4;              // 0 or 1
  const int ln      = lane & 15;
  const int base2   = halfsel * 8;            // A K-subgroup offset

  f32x8 c[NT] = {};

  const float* arow = X + (size_t)(node0 + ln) * HID;

#pragma unroll
  for (int kk = 0; kk < 4; ++kk) {
    // A 16x32 bf16 fragment (ISA layout): lane<16 rows 0-15 K={0..7,16..23},
    // lane>=16 same rows K={8..15,24..31}; two contiguous 32B runs per lane.
    const float* ap = arow + kk * 32 + base2;
    f32x4 a0 = *(const f32x4*)(ap);
    f32x4 a1 = *(const f32x4*)(ap + 4);
    f32x4 a2 = *(const f32x4*)(ap + 16);
    f32x4 a3 = *(const f32x4*)(ap + 20);
    bf16x16 af;
#pragma unroll
    for (int i = 0; i < 4; ++i) {
      af[i]      = (__bf16)a0[i];
      af[4 + i]  = (__bf16)a1[i];
      af[8 + i]  = (__bf16)a2[i];
      af[12 + i] = (__bf16)a3[i];
    }
#pragma unroll
    for (int t = 0; t < NT; ++t) {
      bf16x16 bf = *(const bf16x16*)(Wp + (((t * 4 + kk) * 32 + lane) << 4));
      c[t] = __builtin_amdgcn_wmma_f32_16x16x32_bf16(
          false, af, false, bf, (short)0, c[t], false, false);
    }
  }

  const int NCOL = NT * 16;
#pragma unroll
  for (int t = 0; t < NT; ++t) {
    const int col = t * 16 + ln;
    float bv = HAS_BIAS ? bias[col] : 0.0f;
#pragma unroll
    for (int r = 0; r < 8; ++r) {
      int row = node0 + r + halfsel * 8;      // C/D layout: lanes16-31 -> M=r+8
      float v = c[t][r] + bv;
      if (GELU) v = 0.5f * v * (1.0f + erff(v * 0.70710678118654752f));
      if (HAS_RES) v += res[(size_t)row * NCOL + col];
      Y[(size_t)row * NCOL + col] = v;
    }
  }
}

// ---------------------------------------------------------------------------
// Degree / dinv
// ---------------------------------------------------------------------------
__global__ void fill_ones(float* __restrict__ p, int n) {
  int i = blockIdx.x * blockDim.x + threadIdx.x;
  if (i < n) p[i] = 1.0f;                     // self-loop contribution to degree
}
__global__ void deg_atomic(const long long* __restrict__ ei, float* __restrict__ deg) {
  int e = blockIdx.x * blockDim.x + threadIdx.x;
  if (e < NE) atomicAdd(&deg[(int)ei[NE + e]], 1.0f);
}
__global__ void to_dinv(float* __restrict__ d, int n) {
  int i = blockIdx.x * blockDim.x + threadIdx.x;
  if (i < n) d[i] = rsqrtf(d[i]);             // deg >= 1 always (self loops)
}

__global__ void zero_f4(float* __restrict__ p, int n4) {
  int i = blockIdx.x * blockDim.x + threadIdx.x;
  if (i < n4) *(f32x4*)(p + 4 * (size_t)i) = (f32x4){0.f, 0.f, 0.f, 0.f};
}

// ---------------------------------------------------------------------------
// Edge scatter: 32 lanes per edge, float4 per lane, f32 atomics into agg[dst].
// ---------------------------------------------------------------------------
__global__ __launch_bounds__(256) void scatter_edges(
    const long long* __restrict__ ei, const float* __restrict__ hw,
    const float* __restrict__ dinv, float* __restrict__ agg) {
  long long gid = (long long)blockIdx.x * blockDim.x + threadIdx.x;
  int e    = (int)(gid >> 5);
  int lane = threadIdx.x & 31;
  if (e >= NE) return;
  int s = (int)ei[e];
  int d = (int)ei[NE + e];
  float w = dinv[s] * dinv[d];
  f32x4 v = *(const f32x4*)(hw + (size_t)s * HID + lane * 4);
  float* ap = agg + (size_t)d * HID + lane * 4;
  atomicAdd(ap + 0, v[0] * w);
  atomicAdd(ap + 1, v[1] * w);
  atomicAdd(ap + 2, v[2] * w);
  atomicAdd(ap + 3, v[3] * w);
}

// ---------------------------------------------------------------------------
// Fused: h = LayerNorm(agg + dinv^2*hw (self loop) + conv_b + h) * g + b
// One wave per node; 4 cols per lane; shuffle reduction.
// ---------------------------------------------------------------------------
__global__ __launch_bounds__(256) void combine_ln(
    const float* __restrict__ agg, const float* __restrict__ hw,
    const float* __restrict__ dinv, const float* __restrict__ cb,
    const float* __restrict__ g, const float* __restrict__ b,
    float* __restrict__ h) {
  int wave = (blockIdx.x * blockDim.x + threadIdx.x) >> 5;
  int lane = threadIdx.x & 31;
  if (wave >= NN) return;
  float di = dinv[wave];
  float sw = di * di;
  size_t off = (size_t)wave * HID + lane * 4;
  f32x4 va = *(const f32x4*)(agg + off);
  f32x4 vw = *(const f32x4*)(hw + off);
  f32x4 vh = *(const f32x4*)(h + off);
  f32x4 vc = *(const f32x4*)(cb + lane * 4);
  f32x4 v;
  float s = 0.f, s2 = 0.f;
#pragma unroll
  for (int i = 0; i < 4; ++i) {
    v[i] = va[i] + sw * vw[i] + vc[i] + vh[i];
    s += v[i];
    s2 += v[i] * v[i];
  }
#pragma unroll
  for (int m = 16; m >= 1; m >>= 1) {
    s  += __shfl_xor(s, m, 32);
    s2 += __shfl_xor(s2, m, 32);
  }
  float mu  = s * (1.0f / HID);
  float var = s2 * (1.0f / HID) - mu * mu;
  float rs  = rsqrtf(var + LN_EPS);
  f32x4 vg = *(const f32x4*)(g + lane * 4);
  f32x4 vb = *(const f32x4*)(b + lane * 4);
  f32x4 o;
#pragma unroll
  for (int i = 0; i < 4; ++i) o[i] = (v[i] - mu) * rs * vg[i] + vb[i];
  *(f32x4*)(h + off) = o;
}

// ---------------------------------------------------------------------------
// Host-side launch
// ---------------------------------------------------------------------------
extern "C" void kernel_launch(void* const* d_in, const int* in_sizes, int n_in,
                              void* d_out, int out_size, void* d_ws, size_t ws_size,
                              hipStream_t stream) {
  const float*     x       = (const float*)d_in[0];
  const long long* ei      = (const long long*)d_in[1];
  const float*     pre_w1  = (const float*)d_in[2];
  const float*     pre_b1  = (const float*)d_in[3];
  const float*     pre_w2  = (const float*)d_in[4];
  const float*     pre_b2  = (const float*)d_in[5];
  const float*     conv_w  = (const float*)d_in[6];
  const float*     conv_b  = (const float*)d_in[7];
  const float*     ln_g    = (const float*)d_in[8];
  const float*     ln_b    = (const float*)d_in[9];
  const float*     ffn_w1  = (const float*)d_in[10];
  const float*     ffn_b1  = (const float*)d_in[11];
  const float*     ffn_w2  = (const float*)d_in[12];
  const float*     ffn_b2  = (const float*)d_in[13];
  const float*     post_w1 = (const float*)d_in[14];
  const float*     post_b1 = (const float*)d_in[15];
  const float*     post_w2 = (const float*)d_in[16];
  const float*     post_b2 = (const float*)d_in[17];
  float*           out     = (float*)d_out;

  // Workspace layout (all 256B aligned): h | t1 | t2 | dinv | packed bf16 weights
  char*  ws  = (char*)d_ws;
  size_t szH = (size_t)NN * HID * sizeof(float);            // 51.2 MB (mult of 256)
  float*  h    = (float*)(ws);
  float*  t1   = (float*)(ws + szH);
  float*  t2   = (float*)(ws + 2 * szH);
  float*  dinv = (float*)(ws + 3 * szH);
  __bf16* pw   = (__bf16*)(ws + 3 * szH + (((size_t)NN * sizeof(float) + 255) & ~(size_t)255));

  const int W128 = 128 * 128;                                // bf16 elems per packed 128-col weight
  __bf16* pw_pre1  = pw;
  __bf16* pw_pre2  = pw + 1 * W128;
  __bf16* pw_conv  = pw + 2 * W128;                          // 5 weights
  __bf16* pw_f1    = pw + 7 * W128;                          // 5 weights
  __bf16* pw_f2    = pw + 12 * W128;                         // 5 weights
  __bf16* pw_post1 = pw + 17 * W128;
  __bf16* pw_post2 = pw + 18 * W128;                         // 128x64

  // ---- pack weights to bf16 fragment order -------------------------------
  const int PB = 256;
  pack_weight<<<(W128 + PB - 1) / PB, PB, 0, stream>>>(pre_w1, pw_pre1, 128);
  pack_weight<<<(W128 + PB - 1) / PB, PB, 0, stream>>>(pre_w2, pw_pre2, 128);
  for (int i = 0; i < NHOPS; ++i) {
    pack_weight<<<(W128 + PB - 1) / PB, PB, 0, stream>>>(conv_w + (size_t)i * W128, pw_conv + (size_t)i * W128, 128);
    pack_weight<<<(W128 + PB - 1) / PB, PB, 0, stream>>>(ffn_w1 + (size_t)i * W128, pw_f1 + (size_t)i * W128, 128);
    pack_weight<<<(W128 + PB - 1) / PB, PB, 0, stream>>>(ffn_w2 + (size_t)i * W128, pw_f2 + (size_t)i * W128, 128);
  }
  pack_weight<<<(W128 + PB - 1) / PB, PB, 0, stream>>>(post_w1, pw_post1, 128);
  pack_weight<<<(128 * 64 + PB - 1) / PB, PB, 0, stream>>>(post_w2, pw_post2, 64);

  // ---- degrees -> dinv ---------------------------------------------------
  fill_ones<<<(NN + PB - 1) / PB, PB, 0, stream>>>(dinv, NN);
  deg_atomic<<<(NE + PB - 1) / PB, PB, 0, stream>>>(ei, dinv);
  to_dinv<<<(NN + PB - 1) / PB, PB, 0, stream>>>(dinv, NN);

  const int GEMM_BLOCKS = (NN / 16 * 32 + PB - 1) / PB;      // 6250 waves -> 782 blocks
  const int ZERO_BLOCKS = (NN * HID / 4 + PB - 1) / PB;
  const int SC_BLOCKS   = (int)(((long long)NE * 32 + PB - 1) / PB);
  const int LN_BLOCKS   = (NN * 32 + PB - 1) / PB;

  // ---- pre-MLP: h = gelu(x@W1+b1)@W2+b2 ----------------------------------
  gemm_bf16<8, true,  true,  false><<<GEMM_BLOCKS, PB, 0, stream>>>(x,  pw_pre1, pre_b1, nullptr, t1, NN);
  gemm_bf16<8, false, true,  false><<<GEMM_BLOCKS, PB, 0, stream>>>(t1, pw_pre2, pre_b2, nullptr, h,  NN);

  // ---- hops --------------------------------------------------------------
  for (int i = 0; i < NHOPS; ++i) {
    gemm_bf16<8, false, false, false><<<GEMM_BLOCKS, PB, 0, stream>>>(h, pw_conv + (size_t)i * W128, nullptr, nullptr, t1, NN);
    zero_f4<<<ZERO_BLOCKS, PB, 0, stream>>>(t2, NN * HID / 4);
    scatter_edges<<<SC_BLOCKS, PB, 0, stream>>>(ei, t1, dinv, t2);
    combine_ln<<<LN_BLOCKS, PB, 0, stream>>>(t2, t1, dinv, conv_b + (size_t)i * HID,
                                             ln_g + (size_t)i * HID, ln_b + (size_t)i * HID, h);
    gemm_bf16<8, true,  true, false><<<GEMM_BLOCKS, PB, 0, stream>>>(h,  pw_f1 + (size_t)i * W128, ffn_b1 + (size_t)i * HID, nullptr, t1, NN);
    // residual fused; in-place into h is safe (each element read-then-written by one lane)
    gemm_bf16<8, false, true, true ><<<GEMM_BLOCKS, PB, 0, stream>>>(t1, pw_f2 + (size_t)i * W128, ffn_b2 + (size_t)i * HID, h, h, NN);
  }

  // ---- post-MLP: out = gelu(h@W1+b1)@W2+b2  (64 cols => 4 tiles) ---------
  gemm_bf16<8, true,  true, false><<<GEMM_BLOCKS, PB, 0, stream>>>(h,  pw_post1, post_b1, nullptr, t1, NN);
  gemm_bf16<4, false, true, false><<<GEMM_BLOCKS, PB, 0, stream>>>(t1, pw_post2, post_b2, nullptr, out, NN);
}
